// HouseHolder_11596411699269
// MI455X (gfx1250) — matrix operand order repaired and checked by reading, not compiled, not run on hardware
//
#include <hip/hip_runtime.h>
#include <hip/hip_bf16.h>

#define DIM 512
#define NCOLS 131072

typedef __attribute__((ext_vector_type(16))) __bf16 v16bf;
typedef __attribute__((ext_vector_type(8)))  float  v8f;
typedef __attribute__((ext_vector_type(4)))  unsigned int v4u;
typedef __attribute__((ext_vector_type(8)))  int v8i;
typedef __attribute__((ext_vector_type(4)))  int v4i;

#ifndef __has_builtin
#define __has_builtin(x) 0
#endif
#if __has_builtin(__builtin_amdgcn_tensor_load_to_lds)
#define HAVE_TDM 1
#else
#define HAVE_TDM 0
#endif

union FragBF {
    unsigned int u[8];
    v16bf v;
};

__device__ __forceinline__ unsigned short f2bf(float x) {
    union { float f; unsigned int u; } a; a.f = x;
    unsigned int r = a.u + 0x7FFFu + ((a.u >> 16) & 1u);   // RNE bf16
    return (unsigned short)(r >> 16);
}
__device__ __forceinline__ float bf2f(unsigned short b) {
    union { unsigned int u; float f; } a; a.u = ((unsigned int)b) << 16;
    return a.f;
}

#if HAVE_TDM
// ---------------------------------------------------------------------------
// Issue one TDM 2D tile load: global (strided rows) -> LDS (packed), fp32.
// D# bitfields per CDNA5 ISA ch.8 (group0: count/lds/global/type,
// group1: data_size=4B, tensor dims, tile dims, dim0 stride).
// ---------------------------------------------------------------------------
__device__ __forceinline__ void tdm_load_2d(const void* gptr, unsigned lds_off,
                                            unsigned tile_w, unsigned tile_h,
                                            unsigned tensor_w, unsigned tensor_h,
                                            unsigned row_stride_elems) {
    unsigned long long ga = (unsigned long long)gptr;
    v4u g0;
    g0[0] = 1u;                                                // count = 1
    g0[1] = lds_off;                                           // lds_addr (bytes)
    g0[2] = (unsigned)(ga & 0xFFFFFFFFull);                    // global_addr lo
    g0[3] = (unsigned)((ga >> 32) & 0x01FFFFFFull) | 0x80000000u; // hi | type=2
    v8i g1;
    g1[0] = (int)(2u << 16);                                   // data_size = 4B
    g1[1] = (int)((tensor_w & 0xFFFFu) << 16);                 // dim0 lo16 @63:48
    g1[2] = (int)((tensor_w >> 16) & 0xFFFFu)                  // dim0 hi16 @79:64
          | (int)((tensor_h & 0xFFFFu) << 16);                 // dim1 lo16 @95:80
    g1[3] = (int)((tensor_h >> 16) & 0xFFFFu)                  // dim1 hi16 @111:96
          | (int)(tile_w << 16);                               // tile_dim0 @127:112
    g1[4] = (int)(tile_h & 0xFFFFu);                           // tile_dim1 (tile_dim2=0)
    g1[5] = (int)row_stride_elems;                             // dim0_stride lo32
    g1[6] = 0;                                                 // dim0_stride hi16
    g1[7] = 0;
    v4i z4 = {0, 0, 0, 0};
#if __clang_major__ >= 23
    v8i z8 = {0, 0, 0, 0, 0, 0, 0, 0};
    __builtin_amdgcn_tensor_load_to_lds(g0, g1, z4, z4, z8, 0);
#else
    __builtin_amdgcn_tensor_load_to_lds(g0, g1, z4, z4, 0);
#endif
}
#endif // HAVE_TDM

// ---------------------------------------------------------------------------
// Step 1: column-normalize weights -> Y = V^T, W = -2*Y.  One block per col.
// ---------------------------------------------------------------------------
__global__ void norm_build(const float* __restrict__ w,
                           float* __restrict__ Y, float* __restrict__ W) {
    const int j = blockIdx.x;
    const int t = threadIdx.x;               // 256 threads
    __shared__ float red[256];
    float s = 0.0f;
    for (int i = t; i < DIM; i += 256) {
        float v = w[i * DIM + j];
        s += v * v;
    }
    red[t] = s;
    __syncthreads();
    for (int o = 128; o > 0; o >>= 1) {
        if (t < o) red[t] += red[t + o];
        __syncthreads();
    }
    float nrm = fmaxf(sqrtf(red[0]), 1e-12f);
    float inv = 1.0f / nrm;
    for (int i = t; i < DIM; i += 256) {
        float v = w[i * DIM + j] * inv;
        Y[j * DIM + i] = v;
        W[j * DIM + i] = -2.0f * v;
    }
}

// ---------------------------------------------------------------------------
// WY recursion:  m1[b,i,j] = dot(Y[b*k+i, :], W[b*k+k2+j, :])
// ---------------------------------------------------------------------------
__global__ void m1_kernel(const float* __restrict__ Y, const float* __restrict__ W,
                          float* __restrict__ m1, int k2, int k, int total) {
    int idx = blockIdx.x * 256 + threadIdx.x;
    if (idx >= total) return;
    int per = k2 * k2;
    int b = idx / per;
    int r = idx - b * per;
    int i = r / k2;
    int j = r - i * k2;
    const float* yrow = Y + (size_t)(b * k + i) * DIM;
    const float* wrow = W + (size_t)(b * k + k2 + j) * DIM;
    float s = 0.0f;
#pragma unroll 8
    for (int p = 0; p < DIM; p++) s += yrow[p] * wrow[p];
    m1[idx] = s;
}

// ---------------------------------------------------------------------------
// WY recursion:  W[b*k+k2+j, p] += sum_i m1[b,i,j]*W[b*k+i, p]
// Writes only Wb1 rows, reads only Wb0 rows -> safe in place.
// ---------------------------------------------------------------------------
__global__ void upd_kernel(float* __restrict__ W, const float* __restrict__ m1,
                           int k2, int k) {
    int idx = blockIdx.x * 256 + threadIdx.x;   // < 131072
    int p  = idx & (DIM - 1);
    int rj = idx >> 9;
    int b  = rj / k2;
    int j  = rj - b * k2;
    const float* m1b = m1 + (size_t)b * k2 * k2;
    const float* w0  = W + (size_t)(b * k) * DIM + p;
    float s = 0.0f;
    for (int i = 0; i < k2; i++) s += m1b[i * k2 + j] * w0[(size_t)i * DIM];
    W[(size_t)(b * k + k2 + j) * DIM + p] += s;
}

// ---------------------------------------------------------------------------
// M = I + W^T Y
// ---------------------------------------------------------------------------
__global__ void buildM(const float* __restrict__ W, const float* __restrict__ Y,
                       float* __restrict__ M) {
    int idx = blockIdx.x * 256 + threadIdx.x;    // 512*512 threads
    int i = idx >> 9;
    int j = idx & (DIM - 1);
    float s = (i == j) ? 1.0f : 0.0f;
    for (int kk = 0; kk < DIM; kk++)
        s += W[(size_t)kk * DIM + i] * Y[(size_t)kk * DIM + j];
    M[(size_t)i * DIM + j] = s;
}

// ---------------------------------------------------------------------------
// Big GEMM: out(512 x 131072) = M(512x512) @ X(512x131072), split-bf16 WMMA.
// 256 threads (8 waves); tile BM=128 x BN=128; wave = 16 rows x 8 col-tiles.
// K pipelined in chunks of 32: TDM DMAs the next fp32 tiles into LDS while
// the current step runs 24 WMMAs/wave; conversion to hi/lo bf16 between.
// ---------------------------------------------------------------------------
#define BM  128
#define BN  128
#define KS  32
#define LDA 34   // ushort stride (pad 32 -> 34 to break LDS bank conflicts)

__global__ __launch_bounds__(256)
void gemm_wmma(const float* __restrict__ M, const float* __restrict__ X,
               float* __restrict__ out) {
    __shared__ unsigned short Ah[BM * LDA];
    __shared__ unsigned short Al[BM * LDA];
    __shared__ unsigned short Bh[BN * LDA];   // transposed: [n][k]
    __shared__ unsigned short Bl[BN * LDA];
#if HAVE_TDM
    __shared__ float Af32[BM * KS];           // TDM staging: [row][32] packed
    __shared__ float Xf32[KS * BN];           // TDM staging: [k][128] packed
#endif

    const int m0 = blockIdx.x * BM;   // 4 m-blocks launch-adjacent per n-slab
    const int n0 = blockIdx.y * BN;
    const int t  = threadIdx.x;
    const int wave   = t >> 5;
    const int lane   = t & 31;
    const int laneHi = (lane >> 4) & 1;
    const int lane16 = lane & 15;

    v8f acc[8];
    {
        v8f z = {};
#pragma unroll
        for (int i = 0; i < 8; i++) acc[i] = z;
    }

#if HAVE_TDM
    const unsigned a_off = (unsigned)(size_t)(void*)Af32;  // LDS byte offsets
    const unsigned x_off = (unsigned)(size_t)(void*)Xf32;
    if (wave == 0) {   // one DMA issue per workgroup (EXEC-independent)
        tdm_load_2d(M + (size_t)m0 * DIM, a_off, KS, BM, DIM, DIM, DIM);
        tdm_load_2d(X + n0, x_off, BN, KS, NCOLS, DIM, NCOLS);
    }
#endif

    for (int k0 = 0; k0 < DIM; k0 += KS) {
#if HAVE_TDM
        if (wave == 0) __builtin_amdgcn_s_wait_tensorcnt(0);
        __syncthreads();   // fp32 staging ready; bf16 arrays no longer read
        // --- convert staged fp32 -> hi/lo bf16 LDS ---
#pragma unroll
        for (int i = 0; i < (BM * KS) / 256; i++) {
            int idx = t + i * 256;
            int r = idx >> 5, c = idx & 31;
            float x = Af32[idx];
            unsigned short h = f2bf(x);
            Ah[r * LDA + c] = h;
            Al[r * LDA + c] = f2bf(x - bf2f(h));
        }
#pragma unroll
        for (int i = 0; i < (KS * BN) / 256; i++) {
            int idx = t + i * 256;
            int kk = idx >> 7, c = idx & 127;
            float x = Xf32[idx];
            unsigned short h = f2bf(x);
            Bh[c * LDA + kk] = h;
            Bl[c * LDA + kk] = f2bf(x - bf2f(h));
        }
        __syncthreads();   // bf16 ready; fp32 staging free for next DMA
        if (wave == 0 && k0 + KS < DIM) {
            tdm_load_2d(M + (size_t)m0 * DIM + (k0 + KS), a_off,
                        KS, BM, DIM, DIM, DIM);
            tdm_load_2d(X + (size_t)(k0 + KS) * NCOLS + n0, x_off,
                        BN, KS, NCOLS, DIM, NCOLS);
        }
#else
        // --- fallback: direct global staging with hi/lo bf16 split ---
#pragma unroll
        for (int i = 0; i < (BM * KS) / 256; i++) {
            int idx = t + i * 256;
            int r = idx >> 5, c = idx & 31;
            float x = M[(size_t)(m0 + r) * DIM + (k0 + c)];
            unsigned short h = f2bf(x);
            Ah[r * LDA + c] = h;
            Al[r * LDA + c] = f2bf(x - bf2f(h));
        }
#pragma unroll
        for (int i = 0; i < (KS * BN) / 256; i++) {
            int idx = t + i * 256;
            int kk = idx >> 7, c = idx & 127;
            float x = X[(size_t)(k0 + kk) * NCOLS + (n0 + c)];
            unsigned short h = f2bf(x);
            Bh[c * LDA + kk] = h;
            Bl[c * LDA + kk] = f2bf(x - bf2f(h));
        }
        __syncthreads();
#endif

        // --- A fragments per ISA 16-bit A layout ---
        FragBF a_hi, a_lo;
        const int arow = wave * 16 + lane16;
#pragma unroll
        for (int v = 0; v < 8; v++) {
            int kb = 2 * v + (v >= 4 ? 8 : 0) + (laneHi ? 8 : 0);
            a_hi.u[v] = *(const unsigned int*)&Ah[arow * LDA + kb];
            a_lo.u[v] = *(const unsigned int*)&Al[arow * LDA + kb];
        }

#pragma unroll
        for (int nt = 0; nt < 8; nt++) {
            FragBF b_hi, b_lo;
            const int bcol = nt * 16 + lane16;
#pragma unroll
            for (int v = 0; v < 8; v++) {
                int kb = 2 * v + (v >= 4 ? 8 : 0) + (laneHi ? 8 : 0);
                b_hi.u[v] = *(const unsigned int*)&Bh[bcol * LDA + kb];
                b_lo.u[v] = *(const unsigned int*)&Bl[bcol * LDA + kb];
            }
            acc[nt] = __builtin_amdgcn_wmma_f32_16x16x32_bf16(
                false, a_hi.v, false, b_hi.v, (short)0, acc[nt], false, false);
            acc[nt] = __builtin_amdgcn_wmma_f32_16x16x32_bf16(
                false, a_hi.v, false, b_lo.v, (short)0, acc[nt], false, false);
            acc[nt] = __builtin_amdgcn_wmma_f32_16x16x32_bf16(
                false, a_lo.v, false, b_hi.v, (short)0, acc[nt], false, false);
        }
#if !HAVE_TDM
        __syncthreads();
#endif
    }

    // --- epilogue: C/D layout -> VGPR r is row (r + laneHi*8), lane16 is col ---
    const int mrow = m0 + wave * 16 + laneHi * 8;
#pragma unroll
    for (int nt = 0; nt < 8; nt++) {
        const int ncol = n0 + nt * 16 + lane16;
#pragma unroll
        for (int r = 0; r < 8; r++) {
            out[(size_t)(mrow + r) * NCOLS + ncol] = acc[nt][r];
        }
    }
}

// ---------------------------------------------------------------------------
extern "C" void kernel_launch(void* const* d_in, const int* in_sizes, int n_in,
                              void* d_out, int out_size, void* d_ws, size_t ws_size,
                              hipStream_t stream) {
    (void)in_sizes; (void)n_in; (void)out_size; (void)ws_size;
    const float* X       = (const float*)d_in[0];   // (512, 131072)
    const float* weights = (const float*)d_in[1];   // (512, 512)
    float* out = (float*)d_out;

    char* ws = (char*)d_ws;
    float* Y  = (float*)(ws);                             // 1 MB
    float* W  = (float*)(ws + (1u << 20));                // 1 MB
    float* m1 = (float*)(ws + (2u << 20));                // up to 256 KB
    float* Mm = (float*)(ws + (2u << 20) + (1u << 19));   // 1 MB

    norm_build<<<DIM, 256, 0, stream>>>(weights, Y, W);

    int k = 1;
    for (int s = 0; s < 9; s++) {
        int k2 = k;
        k *= 2;
        int n = DIM / k;
        int tot1 = n * k2 * k2;
        m1_kernel<<<(tot1 + 255) / 256, 256, 0, stream>>>(Y, W, m1, k2, k, tot1);
        upd_kernel<<<(DIM / 2) * DIM / 256, 256, 0, stream>>>(W, m1, k2, k);
    }

    buildM<<<(DIM * DIM) / 256, 256, 0, stream>>>(W, Y, Mm);

    dim3 grid(DIM / BM, NCOLS / BN);   // x = m-block (fast) -> L2 reuse of X slab
    gemm_wmma<<<grid, 256, 0, stream>>>(Mm, X, out);
}
